// ReversiSmallModel_12970801234493
// MI455X (gfx1250) — matrix operand order repaired and checked by reading, not compiled, not run on hardware
//
#include <hip/hip_runtime.h>

typedef __attribute__((ext_vector_type(16))) _Float16 v16h;
typedef __attribute__((ext_vector_type(8)))  _Float16 v8h;
typedef __attribute__((ext_vector_type(4)))  _Float16 v4h;
typedef __attribute__((ext_vector_type(8)))  float    v8f;

#define LPA     128
#define WROWS   384      // 3 * LPA columns in W_in
#define KACT    32       // active features per sample
#define NOUT    30       // layer-stack buckets
#define TILE_M  16       // samples per block (one WMMA M-tile)
#define THREADS 512      // 16 waves of 32

__global__ __launch_bounds__(THREADS)
void reversi_fused_kernel(const int*   __restrict__ indices,
                          const float* __restrict__ values,
                          const int*   __restrict__ ply,
                          const float* __restrict__ W_in,
                          const float* __restrict__ b_in,
                          const float* __restrict__ W_out,
                          const float* __restrict__ b_out,
                          float*       __restrict__ out,
                          int batch)
{
    __shared__ _Float16 x_lds[TILE_M * LPA];   // activated features (f16), row-major
    __shared__ _Float16 w_lds[32 * LPA];       // W_out padded 30->32 rows (f16), row-major
    __shared__ float    o_lds[TILE_M * 32];    // WMMA result tile [16 x 32]
    __shared__ int      ls_lds[TILE_M];        // per-sample output bucket

    const int tid  = threadIdx.x;
    const int wave = tid >> 5;
    const int lane = tid & 31;
    const int s    = blockIdx.x * TILE_M + wave;   // this wave's sample

    // ---- cooperative: W_out (30x128 f32) -> LDS f16, zero-padded to 32 rows ----
    {
        const int e = tid * 8;            // 512 threads * 8 halves = 4096 = 32*128
        const int r = e >> 7;
        const int c = e & 127;
        v8h wv = {};
        if (r < NOUT) {
            const float4 f0 = *(const float4*)(W_out + r * LPA + c);
            const float4 f1 = *(const float4*)(W_out + r * LPA + c + 4);
            wv[0] = (_Float16)f0.x; wv[1] = (_Float16)f0.y;
            wv[2] = (_Float16)f0.z; wv[3] = (_Float16)f0.w;
            wv[4] = (_Float16)f1.x; wv[5] = (_Float16)f1.y;
            wv[6] = (_Float16)f1.z; wv[7] = (_Float16)f1.w;
        }
        *(v8h*)&w_lds[e] = wv;
    }

    // ---- per-wave sparse gather: only the ply-selected 128-wide slice ----
    if (s < batch) {
        const int p  = ply[s];
        const int pa = min(p / 10, 2);            // PA bucket
        const int co = pa * LPA + 4 * lane;       // this lane's 4 columns
        if (lane == 0) ls_lds[wave] = min(p, NOUT - 1);

        const int base  = s * KACT;
        const int   idx_l = indices[base + lane]; // one (idx,val) pair per lane
        const float val_l = values[base + lane];

        float4 acc = make_float4(0.f, 0.f, 0.f, 0.f);

        #pragma unroll 4
        for (int k = 0; k < KACT; ++k) {
            const int   idx = __shfl(idx_l, k, 32);
            const float v   = __shfl(val_l, k, 32);
            // prefetch a future row's cachelines (global_prefetch_b8)
            if (k + 6 < KACT) {
                const int idxp = __shfl(idx_l, k + 6, 32);
                __builtin_prefetch(W_in + (size_t)idxp * WROWS + co, 0, 0);
            }
            const float4 r = *(const float4*)(W_in + (size_t)idx * WROWS + co);
            acc.x += v * r.x; acc.y += v * r.y;
            acc.z += v * r.z; acc.w += v * r.w;
        }

        const float4 bi = *(const float4*)(b_in + co);
        float y[4] = { acc.x + bi.x, acc.y + bi.y, acc.z + bi.z, acc.w + bi.w };

        v4h xv;
        #pragma unroll
        for (int j = 0; j < 4; ++j) {
            const float t = fminf(fmaxf(y[j], 0.f), 1.f);
            xv[j] = (_Float16)(t * t * (1023.0f / 1024.0f));
        }
        *(v4h*)&x_lds[wave * LPA + 4 * lane] = xv;   // one 8-byte DS store
    }
    __syncthreads();

    // ---- wave 0: out_tile[16x32] = x[16x128] @ W_out^T[128x32] via WMMA f16 ----
    if (wave == 0) {
        const int r16   = lane & 15;      // A: row M / B: column N
        const int hi    = lane >> 4;      // lane half select
        const int koffA = hi * 16;        // A halves split K at 16 (contiguous, matches B)
        const int koffB = hi * 16;        // B halves split K at 16 (contiguous, matches A)

        const _Float16* xrow  = &x_lds[r16 * LPA];
        const _Float16* w0row = &w_lds[r16 * LPA];
        const _Float16* w1row = &w_lds[(16 + r16) * LPA];

        v8f c0 = {}; v8f c1 = {};
        #pragma unroll
        for (int kc = 0; kc < 4; ++kc) {
            // A fragment: two contiguous 16B runs -> 2x ds_load_b128
            const v8h alo = *(const v8h*)(xrow + kc * 32 + koffA);
            const v8h ahi = *(const v8h*)(xrow + kc * 32 + 8 + koffA);
            const v16h a  = __builtin_shufflevector(alo, ahi,
                              0, 1, 2, 3, 4, 5, 6, 7, 8, 9, 10, 11, 12, 13, 14, 15);
            // B fragments: 16 contiguous halves (32B) -> 2x ds_load_b128 each
            const v16h b0 = *(const v16h*)(w0row + kc * 32 + koffB);
            const v16h b1 = *(const v16h*)(w1row + kc * 32 + koffB);

            c0 = __builtin_amdgcn_wmma_f32_16x16x32_f16(false, a, false, b0,
                                                        (short)0, c0, false, false);
            c1 = __builtin_amdgcn_wmma_f32_16x16x32_f16(false, a, false, b1,
                                                        (short)0, c1, false, false);
        }
        // C/D layout: VGPR j -> M = j (+8 for lane>=16), N = lane&15 (+tile*16)
        const int mo = hi * 8;
        #pragma unroll
        for (int j = 0; j < 8; ++j) {
            o_lds[(j + mo) * 32 + r16]      = c0[j];
            o_lds[(j + mo) * 32 + 16 + r16] = c1[j];
        }
    }
    __syncthreads();

    // ---- per-sample layer-stack gather ----
    if (tid < TILE_M) {
        const int m = blockIdx.x * TILE_M + tid;
        if (m < batch) {
            const int ls = ls_lds[tid];
            out[m] = o_lds[tid * 32 + ls] + b_out[ls];
        }
    }
}

extern "C" void kernel_launch(void* const* d_in, const int* in_sizes, int n_in,
                              void* d_out, int out_size, void* d_ws, size_t ws_size,
                              hipStream_t stream) {
    // setup_inputs order: indices, values, batch_size, in_features, ply,
    //                     W_in, b_in, W_out, b_out
    const int*   indices = (const int*)  d_in[0];
    const float* values  = (const float*)d_in[1];
    const int*   ply     = (const int*)  d_in[4];
    const float* W_in    = (const float*)d_in[5];
    const float* b_in    = (const float*)d_in[6];
    const float* W_out   = (const float*)d_in[7];
    const float* b_out   = (const float*)d_in[8];
    float*       out     = (float*)d_out;

    const int batch = out_size;                     // output is [B, 1]
    const int grid  = (batch + TILE_M - 1) / TILE_M;

    hipLaunchKernelGGL(reversi_fused_kernel, dim3(grid), dim3(THREADS), 0, stream,
                       indices, values, ply, W_in, b_in, W_out, b_out, out, batch);
}
